// AttentionCache_47459388621297
// MI455X (gfx1250) — matrix operand order repaired and checked by reading, not compiled, not run on hardware
//
#include <hip/hip_runtime.h>
#include <hip/hip_bf16.h>

// ---------------------------------------------------------------------------
// MLA prefill for MI455X (gfx1250): bf16 WMMA everywhere, fp32 accumulate.
// GEMMs use double-buffered LDS staged by GLOBAL_LOAD_ASYNC_TO_LDS_B128
// (ASYNCcnt-tracked).
// Shapes (b=1): S=2048 new tokens, PAST=2048, T=4096, H=16 heads,
// HID=5120, Q_LORA=1536, KV_LORA=512, ROPE=64, NOPE=128, VDIM=128, QH=192.
// ---------------------------------------------------------------------------

#define S_TOK   2048
#define PAST    2048
#define T_TOT   4096
#define NHEAD   16
#define HID     5120
#define QLORA   1536
#define KVLORA  512
#define ROPE_D  64
#define NOPE_D  128
#define VDIM    128
#define QH      192
#define QDIM3   (NHEAD * QH)              // 3072
#define KVDIM   (NHEAD * (NOPE_D + VDIM)) // 4096
#define CTXW    (NHEAD * VDIM)            // 2048

typedef __attribute__((ext_vector_type(16))) __bf16 bf16x16;
typedef __attribute__((ext_vector_type(8)))  __bf16 bf16x8;
typedef __attribute__((ext_vector_type(8)))  float  f32x8;

__device__ __forceinline__ f32x8 wmma_bf16(bf16x16 a, bf16x16 b, f32x8 c) {
  // v_wmma_f32_16x16x32_bf16 : D = A(16x32) * B(32x16) + C(16x16 f32)
  return __builtin_amdgcn_wmma_f32_16x16x32_bf16(false, a, false, b,
                                                 (short)0, c, false, false);
}

// 16-byte global -> LDS copy. Async DMA path on gfx1250 when available
// (no VGPR round-trip, tracked by ASYNCcnt); sync fallback otherwise.
#if defined(__AMDGCN__) && __has_builtin(__builtin_amdgcn_global_load_async_to_lds_b128)
#define HAVE_ASYNC_LDS 1
typedef int v4i __attribute__((vector_size(4 * sizeof(int))));
typedef __attribute__((address_space(1))) v4i as1_v4i;
typedef __attribute__((address_space(3))) v4i as3_v4i;
__device__ __forceinline__ void async_copy16(const void* g, void* l) {
  __builtin_amdgcn_global_load_async_to_lds_b128((as1_v4i*)g, (as3_v4i*)l,
                                                 0, 0);
}
#else
#define HAVE_ASYNC_LDS 0
__device__ __forceinline__ void async_copy16(const void* g, void* l) {
  *(uint4*)l = *(const uint4*)g;   // sync fallback
}
#endif

__device__ __forceinline__ void wait_async_le3() {
#if defined(__AMDGCN__)
  asm volatile("s_wait_asynccnt 0x3" ::: "memory");
#endif
}
__device__ __forceinline__ void wait_async_0() {
#if defined(__AMDGCN__)
  asm volatile("s_wait_asynccnt 0x0" ::: "memory");
#endif
}

// ------------------------------ elementwise --------------------------------

__global__ void k_f32_to_bf16(const float* __restrict__ in,
                              __bf16* __restrict__ out, long n) {
  long i = (long)blockIdx.x * 256 + threadIdx.x;
  if (i < n) out[i] = (__bf16)in[i];
}

// RMSNorm one row per block; writes bf16.
__global__ __launch_bounds__(256)
void k_rmsnorm_bf16(const float* __restrict__ x, int ldx, int width,
                    const float* __restrict__ w,
                    __bf16* __restrict__ y, int ldy) {
  __shared__ float red[8];
  const float* xr = x + (size_t)blockIdx.x * ldx;
  float ss = 0.f;
  for (int i = threadIdx.x; i < width; i += 256) {
    float v = xr[i];
    ss += v * v;
  }
  for (int m = 1; m < 32; m <<= 1) ss += __shfl_xor(ss, m, 32);
  if ((threadIdx.x & 31) == 0) red[threadIdx.x >> 5] = ss;
  __syncthreads();
  float tot = 0.f;
  for (int i = 0; i < 8; ++i) tot += red[i];
  float inv = rsqrtf(tot / (float)width + 1e-6f);
  __bf16* yr = y + (size_t)blockIdx.x * ldy;
  for (int i = threadIdx.x; i < width; i += 256)
    yr[i] = (__bf16)(xr[i] * inv * w[i]);
}

// ------------------------------ GEMM (bf16 WMMA) ---------------------------
// C[M,N] f32 = A[M,K] bf16 @ W[N,K]^T bf16 (+ res_scale * resid).
// 256 thr = 8 waves; block tile 128x64; wave tile 32x32 = 2x2 wmma 16x16x32.
// Double-buffered LDS; each wave issues exactly 3 async b128 ops per stage
// (2 A-rows-of-16 + 1 B-rows-of-8), so "s_wait_asynccnt 3" drains the
// previous stage while the next one is in flight (async loads complete in
// order).
__global__ __launch_bounds__(256)
void k_gemm_bf16(const __bf16* __restrict__ A, const __bf16* __restrict__ B,
                 float* __restrict__ C, int M, int N, int K,
                 const float* __restrict__ resid, float res_scale) {
  __shared__ __align__(32) __bf16 As[2][128][32];
  __shared__ __align__(32) __bf16 Bs[2][64][32];
  const int tid = threadIdx.x;
  const int n0 = blockIdx.x * 64, m0 = blockIdx.y * 128;
  const int w = tid >> 5, lane = tid & 31;
  const int wm = w & 3, wn = w >> 2;
  const int lrow = lane & 15;
  const bool hi = lane >= 16;

  // staging geometry: each lane moves 16B (8 bf16) per async instruction
  const int colv  = (lane & 3) * 8;        // element offset of 16B chunk
  const int arow0 = w * 16 + (lane >> 2);  // wave w covers A rows w*16..+15
  const int brow0 = w * 8 + (lane >> 2);   // wave w covers B rows w*8..+7

  auto stage = [&](int s, int kk) {
    const __bf16* ga0 = A + (size_t)(m0 + arow0) * K + kk + colv;
    async_copy16(ga0, &As[s][arow0][colv]);
    async_copy16(ga0 + (size_t)8 * K, &As[s][arow0 + 8][colv]);
    const __bf16* gb = B + (size_t)(n0 + brow0) * K + kk + colv;
    async_copy16(gb, &Bs[s][brow0][colv]);
  };

  f32x8 acc[2][2] = {};
  stage(0, 0);

  for (int kk = 0; kk < K; kk += 32) {
    const int s = (kk >> 5) & 1;
    const bool more = (kk + 32) < K;
    if (more) {
      stage(s ^ 1, kk + 32);   // next stage in flight while we compute
      wait_async_le3();        // previous (current) stage fully landed
    } else {
      wait_async_0();
    }
    __syncthreads();

    const int ab = hi ? 16 : 0;  // A frag K-chunk base per half-wave
    const int bb = hi ? 16 : 0;  // B frag K base per half-wave
    bf16x16 af[2], bfr[2];
    for (int mt = 0; mt < 2; ++mt) {
      union { bf16x16 v; bf16x8 h8[2]; } u;
      int r = wm * 32 + mt * 16 + lrow;
      u.h8[0] = *(const bf16x8*)&As[s][r][ab];
      u.h8[1] = *(const bf16x8*)&As[s][r][ab + 8];
      af[mt] = u.v;
    }
    for (int nt = 0; nt < 2; ++nt) {
      int r = wn * 32 + nt * 16 + lrow;
      bfr[nt] = *(const bf16x16*)&Bs[s][r][bb];
    }
    for (int mt = 0; mt < 2; ++mt)
      for (int nt = 0; nt < 2; ++nt)
        acc[mt][nt] = wmma_bf16(af[mt], bfr[nt], acc[mt][nt]);

    __syncthreads();  // all waves done reading buf s before it is re-staged
  }

  for (int mt = 0; mt < 2; ++mt)
    for (int nt = 0; nt < 2; ++nt)
      for (int r = 0; r < 8; ++r) {
        int m = m0 + wm * 32 + mt * 16 + (hi ? 8 : 0) + r;
        int n = n0 + wn * 32 + nt * 16 + lrow;
        float v = acc[mt][nt][r];
        if (resid) v += res_scale * resid[(size_t)m * N + n];
        C[(size_t)m * N + n] = v;
      }
}

// ------------------------------ RoPE helpers -------------------------------
// reference: x -> concat(x[0::2], x[1::2]); rotate-half with inv over d/2.
// out[j] = x[2j]*cos - x[2j+1]*sin            (j < 32)
// out[j] = x[2(j-32)+1]*cos + x[2(j-32)]*sin  (j >= 32)
__device__ __forceinline__ float rope_elem(const float* base, int j, int pos) {
  int i = j & 31;
  float inv = __expf(-(float)i * (9.210340371976184f / 32.0f)); // 10000^(-i/32)
  float ang = (float)pos * inv;
  float sn, cs;
  __sincosf(ang, &sn, &cs);
  float xp, rot;
  if (j < 32) { xp = base[2 * j];           rot = -base[2 * j + 1]; }
  else        { int jj = j - 32; xp = base[2 * jj + 1]; rot = base[2 * jj]; }
  return xp * cs + rot * sn;
}

// q (f32 [S][3072]) -> qbf (bf16 [H][S][192]) with RoPE on last 64 dims.
__global__ __launch_bounds__(256)
void k_rope_q(const float* __restrict__ qf, const int* __restrict__ pos,
              __bf16* __restrict__ qbf) {
  int h = blockIdx.x;
  int s = blockIdx.y * 4 + (threadIdx.x >> 6);
  int j = threadIdx.x & 63;
  const float* qr = qf + (size_t)s * QDIM3 + h * QH;
  __bf16* orow = qbf + ((size_t)h * S_TOK + s) * QH;
  orow[2 * j]     = (__bf16)qr[2 * j];
  orow[2 * j + 1] = (__bf16)qr[2 * j + 1];
  orow[NOPE_D + j] = (__bf16)rope_elem(qr + NOPE_D, j, pos[s]);
}

// Build out_k (f32 [H][T][192]) and kbf (bf16 same layout):
// past rows copied; new rows = k_nope from kv + RoPE(k_pe from ckv).
__global__ __launch_bounds__(256)
void k_rope_k_concat(const float* __restrict__ kvf, const float* __restrict__ ckv,
                     const float* __restrict__ past_k, const int* __restrict__ pos,
                     float* __restrict__ out_k, __bf16* __restrict__ kbf) {
  int h = blockIdx.x;
  int t = blockIdx.y * 4 + (threadIdx.x >> 6);
  int j = threadIdx.x & 63;
  float* krow = out_k + ((size_t)h * T_TOT + t) * QH;
  __bf16* kb = kbf + ((size_t)h * T_TOT + t) * QH;
  if (t < PAST) {
    const float* prow = past_k + ((size_t)h * PAST + t) * QH;
    for (int d = j; d < QH; d += 64) {
      float v = prow[d];
      krow[d] = v; kb[d] = (__bf16)v;
    }
  } else {
    int s = t - PAST;
    const float* nrow = kvf + (size_t)s * KVDIM + h * (NOPE_D + VDIM);
    for (int d = j; d < NOPE_D; d += 64) {
      float v = nrow[d];
      krow[d] = v; kb[d] = (__bf16)v;
    }
    const float* pe = ckv + (size_t)s * (KVLORA + ROPE_D) + KVLORA;
    float v = rope_elem(pe, j, pos[s]);
    krow[NOPE_D + j] = v; kb[NOPE_D + j] = (__bf16)v;
  }
}

// out_v (f32 [H][T][128]) + vtb (bf16 transposed [H][128][T]).
__global__ void k_v_concat(const float* __restrict__ kvf,
                           const float* __restrict__ past_v,
                           float* __restrict__ out_v, __bf16* __restrict__ vtb) {
  long idx = (long)blockIdx.x * 256 + threadIdx.x;
  if (idx >= (long)NHEAD * T_TOT * VDIM) return;
  int d = (int)(idx & (VDIM - 1));
  long r = idx >> 7;
  int t = (int)(r & (T_TOT - 1));
  int h = (int)(r >> 12);
  float v = (t < PAST)
      ? past_v[((size_t)h * PAST + t) * VDIM + d]
      : kvf[(size_t)(t - PAST) * KVDIM + h * (NOPE_D + VDIM) + NOPE_D + d];
  out_v[idx] = v;
  vtb[((size_t)h * VDIM + d) * T_TOT + t] = (__bf16)v;
}

// ------------------------------ attention ----------------------------------
// One wave per 16-query tile; online softmax; 12 scores + 8 ctx WMMAs per
// 32-wide T block. K/V per head (2.5 MB) x 16 heads fits in the 192 MB L2,
// so cross-tile re-reads stay on-chip.
__global__ __launch_bounds__(128)
void k_attention(const __bf16* __restrict__ qbf, const __bf16* __restrict__ kbf,
                 const __bf16* __restrict__ vtb, const float* __restrict__ mask,
                 float* __restrict__ ctxf) {
  __shared__ __align__(32) __bf16 Pb[4][16][32];
  const int wid = threadIdx.x >> 5, lane = threadIdx.x & 31;
  const int gt = blockIdx.x * 4 + wid;             // 0..2047
  const int h = gt >> 7;                           // 128 q-tiles per head
  const int q0 = (gt & 127) * 16;
  const int lrow = lane & 15;
  const bool hi = lane >= 16;
  const float sscale = 0.07216878364870323f;       // 192^-0.5

  // Q A-fragments, kept live across the whole T loop (6 chunks of K=32).
  bf16x16 qa[6];
  {
    const __bf16* qr = qbf + ((size_t)h * S_TOK + q0 + lrow) * QH;
    const int b0 = hi ? 16 : 0;
    for (int kc = 0; kc < 6; ++kc) {
      union { bf16x16 v; bf16x8 h8[2]; } u;
      u.h8[0] = *(const bf16x8*)(qr + kc * 32 + b0);
      u.h8[1] = *(const bf16x8*)(qr + kc * 32 + b0 + 8);
      qa[kc] = u.v;
    }
  }

  f32x8 o[8] = {};
  float mrow[8], lsum[8];
  for (int r = 0; r < 8; ++r) { mrow[r] = -1e30f; lsum[r] = 0.f; }

  const int kb0 = hi ? 16 : 0;  // B-fragment K base per half-wave
  for (int tb = 0; tb < T_TOT; tb += 32) {
    f32x8 c0 = {}, c1 = {};
    for (int kc = 0; kc < 6; ++kc) {
      const __bf16* kr = kbf + ((size_t)h * T_TOT + tb + lrow) * QH + kc * 32 + kb0;
      bf16x16 b0 = *(const bf16x16*)kr;
      bf16x16 b1 = *(const bf16x16*)(kr + 16 * QH);
      c0 = wmma_bf16(qa[kc], b0, c0);
      c1 = wmma_bf16(qa[kc], b1, c1);
    }

    const int qb = q0 + (hi ? 8 : 0);
    for (int r = 0; r < 8; ++r) {
      int qq = qb + r;
      c0[r] = c0[r] * sscale + mask[(size_t)qq * T_TOT + tb + lrow];
      c1[r] = c1[r] * sscale + mask[(size_t)qq * T_TOT + tb + 16 + lrow];
    }

    // Online softmax. Half-group (xor 1,2,4,8) row reductions match the
    // C-fragment row mapping, so corr[] lines up with o[] registers directly.
    float corr[8], ps0[8], ps1[8];
    for (int r = 0; r < 8; ++r) {
      float v = fmaxf(c0[r], c1[r]);
      for (int m = 1; m < 16; m <<= 1) v = fmaxf(v, __shfl_xor(v, m, 32));
      float mn = fmaxf(mrow[r], v);
      float cr = __expf(mrow[r] - mn);
      float p0 = __expf(c0[r] - mn);
      float p1 = __expf(c1[r] - mn);
      float rs = p0 + p1;
      for (int m = 1; m < 16; m <<= 1) rs += __shfl_xor(rs, m, 32);
      lsum[r] = lsum[r] * cr + rs;
      mrow[r] = mn; corr[r] = cr; ps0[r] = p0; ps1[r] = p1;
    }

    // Re-stage P through LDS: C-layout -> A-layout for the ctx WMMA.
    const int pr = hi ? 8 : 0;
    for (int r = 0; r < 8; ++r) {
      Pb[wid][pr + r][lrow]      = (__bf16)ps0[r];
      Pb[wid][pr + r][16 + lrow] = (__bf16)ps1[r];
    }
    asm volatile("s_wait_dscnt 0" ::: "memory");  // wave-internal LDS RAW
    union { bf16x16 v; bf16x8 h8[2]; } pu;
    {
      const int b0 = hi ? 16 : 0;
      pu.h8[0] = *(const bf16x8*)&Pb[wid][lrow][b0];
      pu.h8[1] = *(const bf16x8*)&Pb[wid][lrow][b0 + 8];
    }

    for (int dt = 0; dt < 8; ++dt) {
      for (int r = 0; r < 8; ++r) o[dt][r] *= corr[r];
      const __bf16* vr =
          vtb + ((size_t)h * VDIM + dt * 16 + lrow) * T_TOT + tb + kb0;
      bf16x16 bv = *(const bf16x16*)vr;
      o[dt] = wmma_bf16(pu.v, bv, o[dt]);
    }
  }

  const int qb = q0 + (hi ? 8 : 0);
  for (int dt = 0; dt < 8; ++dt) {
    int col = h * VDIM + dt * 16 + lrow;
    for (int r = 0; r < 8; ++r)
      ctxf[(size_t)(qb + r) * CTXW + col] = o[dt][r] / lsum[r];
  }
}

// ------------------------------ launcher -----------------------------------

extern "C" void kernel_launch(void* const* d_in, const int* in_sizes, int n_in,
                              void* d_out, int out_size, void* d_ws, size_t ws_size,
                              hipStream_t stream) {
  (void)in_sizes; (void)n_in; (void)out_size; (void)ws_size;
  const float* hidden   = (const float*)d_in[0];
  const int*   pos      = (const int*)d_in[1];
  const float* mask     = (const float*)d_in[2];
  const float* past_k   = (const float*)d_in[3];
  const float* past_v   = (const float*)d_in[4];
  const float* w_in_ln  = (const float*)d_in[5];
  const float* w_qa     = (const float*)d_in[6];
  const float* w_qa_ln  = (const float*)d_in[7];
  const float* w_qb     = (const float*)d_in[8];
  const float* w_kva    = (const float*)d_in[9];
  const float* w_kva_ln = (const float*)d_in[10];
  const float* w_kvb    = (const float*)d_in[11];
  const float* w_o      = (const float*)d_in[12];

  float* out   = (float*)d_out;
  float* out_k = out + (size_t)S_TOK * HID;
  float* out_v = out_k + (size_t)NHEAD * T_TOT * QH;

  // workspace carve-out (~241 MB)
  char* base = (char*)d_ws;
  size_t off = 0;
  auto alloc = [&](size_t bytes) -> char* {
    char* p = base + off;
    off = (off + bytes + 255) & ~(size_t)255;
    return p;
  };
  __bf16* xb     = (__bf16*)alloc((size_t)S_TOK * HID * 2);
  __bf16* wqa_b  = (__bf16*)alloc((size_t)QLORA * HID * 2);
  __bf16* wqb_b  = (__bf16*)alloc((size_t)QDIM3 * QLORA * 2);
  __bf16* wkva_b = (__bf16*)alloc((size_t)(KVLORA + ROPE_D) * HID * 2);
  __bf16* wkvb_b = (__bf16*)alloc((size_t)KVDIM * KVLORA * 2);
  __bf16* wo_b   = (__bf16*)alloc((size_t)HID * CTXW * 2);
  float*  qa     = (float*) alloc((size_t)S_TOK * QLORA * 4);
  __bf16* qan    = (__bf16*)alloc((size_t)S_TOK * QLORA * 2);
  float*  qf     = (float*) alloc((size_t)S_TOK * QDIM3 * 4);
  __bf16* qbf    = (__bf16*)alloc((size_t)NHEAD * S_TOK * QH * 2);
  float*  ckv    = (float*) alloc((size_t)S_TOK * (KVLORA + ROPE_D) * 4);
  __bf16* ckvn   = (__bf16*)alloc((size_t)S_TOK * KVLORA * 2);
  float*  kvf    = (float*) alloc((size_t)S_TOK * KVDIM * 4);
  __bf16* kbf    = (__bf16*)alloc((size_t)NHEAD * T_TOT * QH * 2);
  __bf16* vtb    = (__bf16*)alloc((size_t)NHEAD * VDIM * T_TOT * 2);
  float*  ctxf   = (float*) alloc((size_t)S_TOK * CTXW * 4);
  __bf16* ctxb   = (__bf16*)alloc((size_t)S_TOK * CTXW * 2);

  auto cvt = [&](const float* src, __bf16* dst, long n) {
    k_f32_to_bf16<<<dim3((unsigned)((n + 255) / 256)), 256, 0, stream>>>(src, dst, n);
  };
  cvt(w_qa,  wqa_b,  (long)QLORA * HID);
  cvt(w_qb,  wqb_b,  (long)QDIM3 * QLORA);
  cvt(w_kva, wkva_b, (long)(KVLORA + ROPE_D) * HID);
  cvt(w_kvb, wkvb_b, (long)KVDIM * KVLORA);
  cvt(w_o,   wo_b,   (long)HID * CTXW);

  // x = rmsnorm(hidden, w_in_ln)
  k_rmsnorm_bf16<<<S_TOK, 256, 0, stream>>>(hidden, HID, HID, w_in_ln, xb, HID);

  // q_a = x @ w_qa^T ; rmsnorm ; q = @ w_qb^T
  k_gemm_bf16<<<dim3(QLORA / 64, S_TOK / 128), 256, 0, stream>>>(
      xb, wqa_b, qa, S_TOK, QLORA, HID, nullptr, 0.f);
  k_rmsnorm_bf16<<<S_TOK, 256, 0, stream>>>(qa, QLORA, QLORA, w_qa_ln, qan, QLORA);
  k_gemm_bf16<<<dim3(QDIM3 / 64, S_TOK / 128), 256, 0, stream>>>(
      qan, wqb_b, qf, S_TOK, QDIM3, QLORA, nullptr, 0.f);

  // ckv = x @ w_kva^T ; rmsnorm(c_kv) ; kv = @ w_kvb^T
  k_gemm_bf16<<<dim3((KVLORA + ROPE_D) / 64, S_TOK / 128), 256, 0, stream>>>(
      xb, wkva_b, ckv, S_TOK, KVLORA + ROPE_D, HID, nullptr, 0.f);
  k_rmsnorm_bf16<<<S_TOK, 256, 0, stream>>>(ckv, KVLORA + ROPE_D, KVLORA,
                                            w_kva_ln, ckvn, KVLORA);
  k_gemm_bf16<<<dim3(KVDIM / 64, S_TOK / 128), 256, 0, stream>>>(
      ckvn, wkvb_b, kvf, S_TOK, KVDIM, KVLORA, nullptr, 0.f);

  // RoPE + cache concat (also emits fp32 k/v outputs)
  k_rope_q<<<dim3(NHEAD, S_TOK / 4), 256, 0, stream>>>(qf, pos, qbf);
  k_rope_k_concat<<<dim3(NHEAD, T_TOT / 4), 256, 0, stream>>>(
      kvf, ckv, past_k, pos, out_k, kbf);
  {
    long n = (long)NHEAD * T_TOT * VDIM;
    k_v_concat<<<dim3((unsigned)((n + 255) / 256)), 256, 0, stream>>>(
        kvf, past_v, out_v, vtb);
  }

  // attention: 16 heads x 128 q-tiles, 4 waves/block
  k_attention<<<NHEAD * (S_TOK / 16) / 4, 128, 0, stream>>>(
      qbf, kbf, vtb, mask, ctxf);

  // out = hidden/8 + ctx @ w_o^T
  cvt(ctxf, ctxb, (long)S_TOK * CTXW);
  k_gemm_bf16<<<dim3(HID / 64, S_TOK / 128), 256, 0, stream>>>(
      ctxb, wo_b, out, S_TOK, HID, CTXW, hidden, 0.125f);
}
